// SPP_CNN_2869038153933
// MI455X (gfx1250) — matrix operand order repaired and checked by reading, not compiled, not run on hardware
//
#include <hip/hip_runtime.h>
#include <hip/hip_bf16.h>

#define C_CH 64
#define T_IN 16384
#define B_N 32
#define KW 21
#define TC (T_IN - KW + 1)   // 16364
#define NCHUNK 42            // (C_CH*KW)/32 K-chunks of 32
#define TBLK 128             // output t per block
#define NROW (TBLK + KW - 1) // 148 staged rows
#define EPS 1e-5f
#define NEG 0.01f
#define NFEAT 384

typedef __attribute__((ext_vector_type(16))) __bf16          v16bf;
typedef __attribute__((ext_vector_type(16))) unsigned short  v16u;
typedef __attribute__((ext_vector_type(8)))  float           v8f;
typedef __attribute__((ext_vector_type(4)))  int             v4i;

// ---- async global->LDS path (CDNA5), with graceful fallback -------------
#if defined(__has_builtin)
# if __has_builtin(__builtin_amdgcn_global_load_async_to_lds_b128)
#  define USE_ASYNC 1
# endif
#endif
#ifndef USE_ASYNC
# define USE_ASYNC 0
#endif

typedef __attribute__((address_space(1))) v4i g_v4i;   // global 128b payload ptr
typedef __attribute__((address_space(3))) v4i l_v4i;   // LDS    128b payload ptr

__device__ __forceinline__ unsigned short f2bf(float f) {
  unsigned u = __float_as_uint(f);
  u += 0x7FFFu + ((u >> 16) & 1u);     // round-to-nearest-even
  return (unsigned short)(u >> 16);
}
__device__ __forceinline__ float bf2f(unsigned short h) {
  return __uint_as_float(((unsigned)h) << 16);
}

__device__ __forceinline__ v8f wmma_bf16(v16u a, v16u b, v8f c) {
  return __builtin_amdgcn_wmma_f32_16x16x32_bf16(
      false, __builtin_bit_cast(v16bf, a),
      false, __builtin_bit_cast(v16bf, b),
      (short)0, c, false, false);
}

// ---------------- 1) convert x: fp32 [b][ci][t] -> bf16 [b][t][ci] -------
__global__ __launch_bounds__(256) void k_cvt_x_t(const float* __restrict__ x,
                                                 unsigned short* __restrict__ xt) {
  __shared__ __align__(16) unsigned short tile[64 * 64];
  int b  = blockIdx.y;
  int t0 = blockIdx.x * 64;
  int tid = threadIdx.x;
  const float* xp = x + b * (C_CH * T_IN);
  for (int i = tid; i < 64 * 64; i += 256) {
    int ci = i >> 6, tt = i & 63;                     // coalesced over tt
    tile[tt * 64 + ci] = f2bf(xp[ci * T_IN + t0 + tt]);
  }
  __syncthreads();
  uint4* dst = (uint4*)(xt + (b * T_IN + t0) * C_CH); // contiguous slab out
  const uint4* src = (const uint4*)tile;
  for (int i = tid; i < 512; i += 256)                // 8 KB tile
    dst[i] = src[i];
}

// ---------------- 2) pack weights into WMMA A-fragment layout ------------
// A is 16x32 bf16 (M=co tile, K=32 ci of one (k, half) chunk).
// Contiguous K order: element j of lane maps to K = j + (lane>=16 ? 16 : 0).
// Flat: wf[ ((ct*NCHUNK + c)*32 + lane)*16 + j ]  (32B contiguous per lane).
__global__ void k_pack_w(const float* __restrict__ w, unsigned short* __restrict__ wf) {
  int idx = blockIdx.x * 256 + threadIdx.x;           // 4*NCHUNK*32*16 = 86016
  if (idx >= 4 * NCHUNK * 32 * 16) return;
  int j    = idx & 15;
  int lane = (idx >> 4) & 31;
  int c    = (idx >> 9) % NCHUNK;
  int ct   = (idx >> 9) / NCHUNK;
  int k = c >> 1, h = c & 1;
  int K = j + ((lane >= 16) ? 16 : 0);
  int ci = h * 32 + K;
  int co = ct * 16 + (lane & 15);
  wf[idx] = f2bf(w[(co * C_CH + ci) * KW + k]);
}

// ---------------- 3) conv as implicit GEMM via bf16 WMMA -----------------
// Block: 256 threads = 8 waves; covers co 0..63 x 128 output t.
// Wave: 32co x 32t = 4 accumulators -> each A/B fragment reused twice.
// Staging: contiguous 148x128B slab, async b128 copies into LDS [tt][ci].
__global__ __launch_bounds__(256) void k_conv(const unsigned short* __restrict__ xt,
                                              const unsigned short* __restrict__ wf,
                                              unsigned short* __restrict__ y) {
  __shared__ __align__(32) unsigned short xs[NROW * 64];  // [tt][ci], 18944 B
  int b  = blockIdx.y;
  int t0 = blockIdx.x * TBLK;
  int tid = threadIdx.x;

  const unsigned short* slab = xt + (b * T_IN + t0) * C_CH;
  for (int q = tid; q < NROW * 8; q += 256) {             // 16B chunks
    int row = q >> 3;
    int gr = (t0 + row > T_IN - 1) ? (T_IN - 1 - t0) : row;  // edge clamp
    const unsigned short* gsrc = slab + gr * 64 + (q & 7) * 8;
#if USE_ASYNC
    __builtin_amdgcn_global_load_async_to_lds_b128(
        (g_v4i*)gsrc, (l_v4i*)&xs[q * 8], 0, 0);
#else
    *(v16u*)&xs[q * 8] = *(const v16u*)gsrc;
#endif
  }
#if USE_ASYNC
# if __has_builtin(__builtin_amdgcn_s_wait_asynccnt)
  __builtin_amdgcn_s_wait_asynccnt(0);
# else
  asm volatile("s_wait_asynccnt 0" ::: "memory");
# endif
#endif
  __syncthreads();

  int wave = tid >> 5, lane = tid & 31;
  int wc = wave & 1;            // co group (32 channels)
  int wt = wave >> 1;           // t group (32 outputs)
  int n  = lane & 15;
  int kb = (lane >> 4) << 4;    // B-layout K base: 0 or 16

  v8f acc[2][2] = {{{}, {}}, {{}, {}}};   // [co subtile][t subtile]
  const v16u* wfa0 = (const v16u*)wf + (wc * 2 + 0) * (NCHUNK * 32) + lane;
  const v16u* wfa1 = (const v16u*)wf + (wc * 2 + 1) * (NCHUNK * 32) + lane;
  int tb = wt * 32 + n;

  for (int c = 0; c < NCHUNK; ++c) {
    int k = c >> 1, h = c & 1;
    v16u a0 = wfa0[c * 32];
    v16u a1 = wfa1[c * 32];
    int base = h * 32 + kb;
    v16u b0 = *(const v16u*)&xs[(tb + k) * 64 + base];        // max row 147
    v16u b1 = *(const v16u*)&xs[(tb + 16 + k) * 64 + base];
    acc[0][0] = wmma_bf16(a0, b0, acc[0][0]);
    acc[0][1] = wmma_bf16(a0, b1, acc[0][1]);
    acc[1][0] = wmma_bf16(a1, b0, acc[1][0]);
    acc[1][1] = wmma_bf16(a1, b1, acc[1][1]);
  }

  // C/D layout: VGPR r -> M = r + (lane>=16 ? 8 : 0), N = lane&15
  int m0 = (lane >> 4) << 3;
#pragma unroll
  for (int s = 0; s < 2; ++s) {
    int t = t0 + wt * 32 + s * 16 + n;
    if (t < TC) {
#pragma unroll
      for (int ct = 0; ct < 2; ++ct) {
        unsigned short* yp = y + (b * C_CH + wc * 32 + ct * 16 + m0) * TC + t;
#pragma unroll
        for (int r = 0; r < 8; ++r)
          yp[r * TC] = f2bf(acc[ct][s][r]);
      }
    }
  }
}

// ---------------- 4) BN train stats -> folded scale/shift ----------------
__global__ void k_bnstat(const unsigned short* __restrict__ y,
                         const float* __restrict__ gamma, const float* __restrict__ beta,
                         float* __restrict__ scale, float* __restrict__ shift) {
  int c = blockIdx.x, tid = threadIdx.x;
  float s = 0.f, s2 = 0.f;
  for (int b = 0; b < B_N; ++b) {
    const unsigned short* yp = y + (b * C_CH + c) * TC;
    for (int t = tid; t < TC; t += 256) {
      float v = bf2f(yp[t]);
      s += v; s2 += v * v;
    }
  }
  __shared__ float rs[256], rq[256];
  rs[tid] = s; rq[tid] = s2;
  __syncthreads();
  for (int o = 128; o > 0; o >>= 1) {
    if (tid < o) { rs[tid] += rs[tid + o]; rq[tid] += rq[tid + o]; }
    __syncthreads();
  }
  if (tid == 0) {
    float inv  = 1.f / (float)(B_N * TC);
    float mean = rs[0] * inv;
    float var  = rq[0] * inv - mean * mean;
    float sc   = rsqrtf(var + EPS) * gamma[c];
    scale[c] = sc;
    shift[c] = beta[c] - mean * sc;
  }
}

// ---------------- 5) SPP (max branch + avg branch) -----------------------
__device__ __forceinline__ float blk_reduce(float v, float* buf, int tid, bool isMax) {
  buf[tid] = v; __syncthreads();
  for (int o = 128; o > 0; o >>= 1) {
    if (tid < o) buf[tid] = isMax ? fmaxf(buf[tid], buf[tid + o]) : (buf[tid] + buf[tid + o]);
    __syncthreads();
  }
  float r = buf[0]; __syncthreads();
  return r;
}

__global__ void k_spp(const unsigned short* __restrict__ y1, const unsigned short* __restrict__ y2,
                      const float* __restrict__ bn, const int* __restrict__ orig_len,
                      float* __restrict__ feat) {
  int c = blockIdx.x, b = blockIdx.y, tid = threadIdx.x;
  int L = orig_len[b] - (KW - 1);
  int stride1 = L >> 1, kern1 = (L + 1) >> 1;          // floor + ceil = L
  float s1 = bn[c], h1 = bn[64 + c], s2 = bn[128 + c], h2 = bn[192 + c];
  const unsigned short* p1 = y1 + (b * C_CH + c) * TC;
  const unsigned short* p2 = y2 + (b * C_CH + c) * TC;

  const float NINF = -3.402823466e38f;
  float m0 = NINF, m10 = NINF, m11 = NINF;
  float a0 = 0.f, a10 = 0.f, a11 = 0.f;
  for (int t = tid; t < L; t += 256) {
    float v1 = s1 * bf2f(p1[t]) + h1;  v1 = v1 >= 0.f ? v1 : NEG * v1;
    float v2 = s2 * bf2f(p2[t]) + h2;  v2 = v2 >= 0.f ? v2 : NEG * v2;
    m0 = fmaxf(m0, v1);
    a0 += v2;
    if (t < kern1)    { m10 = fmaxf(m10, v1); a10 += v2; }
    if (t >= stride1) { m11 = fmaxf(m11, v1); a11 += v2; }
  }
  __shared__ float buf[256];
  m0  = blk_reduce(m0,  buf, tid, true);
  m10 = blk_reduce(m10, buf, tid, true);
  m11 = blk_reduce(m11, buf, tid, true);
  a0  = blk_reduce(a0,  buf, tid, false);
  a10 = blk_reduce(a10, buf, tid, false);
  a11 = blk_reduce(a11, buf, tid, false);
  if (tid == 0) {
    float* fb = feat + b * NFEAT;
    fb[c]               = m0;                       // p1 level0
    fb[64 + 2 * c + 0]  = m10;                      // p1 level1
    fb[64 + 2 * c + 1]  = m11;
    fb[192 + c]         = a0  / (float)L;           // p2 level0 (kern = L)
    fb[256 + 2 * c + 0] = a10 / (float)kern1;       // p2 level1
    fb[256 + 2 * c + 1] = a11 / (float)kern1;
  }
}

// ---------------- 6) FC --------------------------------------------------
__global__ void k_fc(const float* __restrict__ feat, const float* __restrict__ fcw,
                     const float* __restrict__ fcb, float* __restrict__ out) {
  int i = threadIdx.x;
  if (i < B_N * 2) {
    int b = i >> 1, task = i & 1;
    float acc = fcb[task];
    for (int f = 0; f < NFEAT; ++f)
      acc += feat[b * NFEAT + f] * fcw[task * NFEAT + f];
    out[i] = acc;
  }
}

extern "C" void kernel_launch(void* const* d_in, const int* in_sizes, int n_in,
                              void* d_out, int out_size, void* d_ws, size_t ws_size,
                              hipStream_t stream) {
  const float* x        = (const float*)d_in[0];
  const int*   orig_len = (const int*)d_in[1];
  const float* w1       = (const float*)d_in[2];
  const float* g1       = (const float*)d_in[3];
  const float* b1       = (const float*)d_in[4];
  const float* w2       = (const float*)d_in[5];
  const float* g2       = (const float*)d_in[6];
  const float* b2       = (const float*)d_in[7];
  const float* fcw      = (const float*)d_in[8];
  const float* fcb      = (const float*)d_in[9];
  float* out = (float*)d_out;

  char* ws = (char*)d_ws;
  size_t off = 0;
  auto alloc = [&](size_t bytes) -> void* {
    void* p = ws + off;
    off = (off + bytes + 255) & ~(size_t)255;
    return p;
  };
  unsigned short* xt  = (unsigned short*)alloc((size_t)B_N * C_CH * T_IN * 2);
  unsigned short* wf1 = (unsigned short*)alloc((size_t)4 * NCHUNK * 32 * 16 * 2);
  unsigned short* wf2 = (unsigned short*)alloc((size_t)4 * NCHUNK * 32 * 16 * 2);
  unsigned short* y1  = (unsigned short*)alloc((size_t)B_N * C_CH * TC * 2);
  unsigned short* y2  = (unsigned short*)alloc((size_t)B_N * C_CH * TC * 2);
  float* bnp  = (float*)alloc(4 * 64 * sizeof(float));   // scale1,shift1,scale2,shift2
  float* feat = (float*)alloc((size_t)B_N * NFEAT * sizeof(float));

  dim3 gcvt(T_IN / 64, B_N);
  k_cvt_x_t<<<gcvt, 256, 0, stream>>>(x, xt);
  k_pack_w<<<(4 * NCHUNK * 32 * 16) / 256, 256, 0, stream>>>(w1, wf1);
  k_pack_w<<<(4 * NCHUNK * 32 * 16) / 256, 256, 0, stream>>>(w2, wf2);

  dim3 gconv((TC + TBLK - 1) / TBLK, B_N);
  k_conv<<<gconv, 256, 0, stream>>>(xt, wf1, y1);
  k_conv<<<gconv, 256, 0, stream>>>(xt, wf2, y2);

  k_bnstat<<<C_CH, 256, 0, stream>>>(y1, g1, b1, bnp, bnp + 64);
  k_bnstat<<<C_CH, 256, 0, stream>>>(y2, g2, b2, bnp + 128, bnp + 192);

  dim3 gspp(C_CH, B_N);
  k_spp<<<gspp, 256, 0, stream>>>(y1, y2, bnp, orig_len, feat);

  k_fc<<<1, 64, 0, stream>>>(feat, fcw, fcb, out);
}
